// Decoder_10204842295482
// MI455X (gfx1250) — matrix operand order repaired and checked by reading, not compiled, not run on hardware
//
#include <hip/hip_runtime.h>

typedef __attribute__((ext_vector_type(16))) _Float16 v16h;
typedef __attribute__((ext_vector_type(8)))  _Float16 v8h;
typedef __attribute__((ext_vector_type(8)))  float    v8f;

#define B_    256
#define H_    256
#define T_    512
#define O_    256
#define G4H   1024
#define NBLK  16     // one block (col-tile) per WGP
#define TPBL  512    // lstm: 16 waves = 16 M-tiles
#define TPBF  256    // fc: 8 waves
#define BPAD  264    // 256 + 8 halves pad -> row stride 132 dwords == 4 mod 64 banks

// ---------------------------------------------------------------- helpers

__device__ __forceinline__ v8f wmma16(v16h a, v16h b, v8f c) {
  // (neg_a, A, neg_b, B, c_mod, C, reuse_a, reuse_b)
  return __builtin_amdgcn_wmma_f32_16x16x32_f16(false, a, false, b, (short)0, c, false, false);
}

__device__ __forceinline__ float sigm_(float x) {
  return 1.0f / (1.0f + __expf(-x));
}
__device__ __forceinline__ float tanh_(float x) {
  float ax = fabsf(x);
  float e  = __expf(2.0f * ax);
  float t  = 1.0f - 2.0f / (e + 1.0f);   // safe for large |x|
  return copysignf(t, x);
}

// A-operand fragment for v_wmma_f32_16x16x32_f16 (ISA 7.12.2 16-bit A 16x32):
// lane L: row = base + (L&15); halves 0-15 = K[k0 + (L>>4)*16 ..] (contiguous, matches B)
__device__ __forceinline__ v16h load_a_frag(const _Float16* __restrict__ Ar, int k0, int hi) {
  v8h alo = *(const v8h*)(Ar + k0 + hi * 16);
  v8h ahi = *(const v8h*)(Ar + k0 + hi * 16 + 8);
  return __builtin_shufflevector(alo, ahi, 0,1,2,3,4,5,6,7,8,9,10,11,12,13,14,15);
}

// gates(acc[g]) += A(mtile rows, K=256) x W^T   for the 4 gate tiles of col-tile ct.
// B from GLOBAL (used once in the prologue for the hoisted x @ W_ih0).
__device__ __forceinline__ void mm4_globalB(v8f acc[4], const _Float16* __restrict__ A,
                                            const _Float16* __restrict__ W,
                                            int mtile, int ct, int lane) {
  const int nlo = lane & 15;
  const int hi  = lane >> 4;
  const _Float16* Ar = A + (mtile * 16 + nlo) * H_;
  __builtin_prefetch(Ar, 0, 1);
#pragma unroll
  for (int kb = 0; kb < H_ / 32; ++kb) {
    const int k0 = kb * 32;
    v16h a = load_a_frag(Ar, k0, hi);
    const int kB = k0 + hi * 16;       // B: lane L holds N=(L&15), K = kB..kB+15
#pragma unroll
    for (int g = 0; g < 4; ++g) {
      v16h b = *(const v16h*)(W + (g * 256 + ct * 16 + nlo) * H_ + kB);
      acc[g] = wmma16(a, b, acc[g]);
    }
  }
}

// Same, but B staged in LDS: Bs[g*16 + n][k], row stride BPAD halves.
__device__ __forceinline__ void mm4_ldsB(v8f acc[4], const _Float16* __restrict__ A,
                                         const _Float16 (*Bs)[BPAD],
                                         int mtile, int lane) {
  const int nlo = lane & 15;
  const int hi  = lane >> 4;
  const _Float16* Ar = A + (mtile * 16 + nlo) * H_;
  __builtin_prefetch(Ar, 0, 1);
#pragma unroll
  for (int kb = 0; kb < H_ / 32; ++kb) {
    const int k0 = kb * 32;
    v16h a = load_a_frag(Ar, k0, hi);
    const int kB = k0 + hi * 16;
#pragma unroll
    for (int g = 0; g < 4; ++g) {
      v16h b = *(const v16h*)&Bs[g * 16 + nlo][kB];   // ds_load_b128 x2, 2-cycle banks
      acc[g] = wmma16(a, b, acc[g]);
    }
  }
}

__device__ __forceinline__ void grid_barrier(unsigned* cnt, unsigned epoch) {
  __syncthreads();                       // all waves' stores issued & waited
  if (threadIdx.x == 0) {
    __threadfence();                     // device-scope release of our stores
    __hip_atomic_fetch_add(cnt, 1u, __ATOMIC_ACQ_REL, __HIP_MEMORY_SCOPE_AGENT);
    while (__hip_atomic_load(cnt, __ATOMIC_ACQUIRE, __HIP_MEMORY_SCOPE_AGENT) <
           epoch * (unsigned)NBLK) {
      __builtin_amdgcn_s_sleep(1);
    }
    __threadfence();                     // acquire: invalidate stale WGP$ lines
  }
  __syncthreads();
}

// ---------------------------------------------------------------- prep kernels

__global__ void cvt_f16_kernel(_Float16* dst, const float* src, int n) {
  int i = blockIdx.x * blockDim.x + threadIdx.x;
  if (i < n) dst[i] = (_Float16)src[i];
}

__global__ void init_barrier_kernel(unsigned* barc) {
  if (threadIdx.x == 0 && blockIdx.x == 0) *barc = 0u;  // ws poisoned 0xAA each run
}

// ---------------------------------------------------------------- persistent LSTM
//
// Grid: 16 blocks (= col-tile ct = hidden cols ct*16..+16) x 512 threads (16 waves,
// wave = M-tile). Per step: ONE grid barrier (between layer0 and layer1).
// B operands (Wh0, Wi1, Wh1 gate tiles for this ct) live in LDS for all 512 steps.
// Z0 = x@Wih0^T + b (hoisted) and both c states live entirely in registers.

__global__ __launch_bounds__(TPBL) void lstm_kernel(
    const _Float16* __restrict__ xh,
    const _Float16* __restrict__ Wi0, const _Float16* __restrict__ Wh0,
    const _Float16* __restrict__ Wi1, const _Float16* __restrict__ Wh1,
    const float* __restrict__ bih, const float* __restrict__ bhh,
    const float* __restrict__ c0in, const float* __restrict__ c1in,
    _Float16* h0A, _Float16* h0B, _Float16* h1A, _Float16* h1B,
    _Float16* __restrict__ seq, unsigned* barc) {
  __shared__ _Float16 Bsm[3][64][BPAD];   // 101376 B: [Wh0|Wi1|Wh1][4 gates x 16 rows][K]

  const int tid   = threadIdx.x;
  const int lane  = tid & 31;
  const int mtile = tid >> 5;            // 0..15
  const int ct    = blockIdx.x;          // 0..15
  const int nlo   = lane & 15;
  const int hi    = lane >> 4;
  const int rbase = mtile * 16 + hi * 8; // C layout: VGPR r -> row rbase+r, col = ct*16+nlo
  const int col   = ct * 16 + nlo;

  // ---- stage B tiles into LDS (once) ----
  for (int ch = tid; ch < 3 * 64 * (H_ / 8); ch += TPBL) {
    const int m3  = ch / (64 * 32);
    const int rem = ch % (64 * 32);
    const int row = rem >> 5;            // g*16 + r
    const int kc  = (rem & 31) * 8;
    const int g   = row >> 4, rr = row & 15;
    const _Float16* W = (m3 == 0) ? Wh0 : (m3 == 1) ? Wi1 : Wh1;
    *(v8h*)&Bsm[m3][row][kc] = *(const v8h*)(W + (g * 256 + ct * 16 + rr) * H_ + kc);
  }

  // ---- hoisted Z0 = x @ Wih0^T + b_ih0 + b_hh0 (stays in registers) ----
  v8f z0[4];
#pragma unroll
  for (int g = 0; g < 4; ++g)
#pragma unroll
    for (int r = 0; r < 8; ++r) z0[g][r] = 0.0f;
  mm4_globalB(z0, xh, Wi0, mtile, ct, lane);

  float z1b[4];
#pragma unroll
  for (int g = 0; g < 4; ++g) {
    const int n = g * 256 + col;
    const float b0 = bih[n] + bhh[n];
#pragma unroll
    for (int r = 0; r < 8; ++r) z0[g][r] += b0;
    z1b[g] = bih[G4H + n] + bhh[G4H + n];
  }

  // ---- cell states in registers (c is not an output) ----
  float c0r[8], c1r[8];
#pragma unroll
  for (int r = 0; r < 8; ++r) {
    c0r[r] = c0in[(rbase + r) * H_ + col];
    c1r[r] = c1in[(rbase + r) * H_ + col];
  }

  __syncthreads();                        // LDS B ready

  _Float16 *h0c = h0A, *h0n = h0B, *h1c = h1A, *h1n = h1B;

  for (int t = 0; t < T_; ++t) {
    // -------- layer 0: gates = Z0 + h0 @ Wh0^T --------
    v8f acc[4];
#pragma unroll
    for (int g = 0; g < 4; ++g) acc[g] = z0[g];
    mm4_ldsB(acc, h0c, Bsm[0], mtile, lane);

#pragma unroll
    for (int r = 0; r < 8; ++r) {
      const float ig = sigm_(acc[0][r]);
      const float fg = sigm_(acc[1][r]);
      const float gg = tanh_(acc[2][r]);
      const float og = sigm_(acc[3][r]);
      c0r[r] = fg * c0r[r] + ig * gg;
      h0n[(rbase + r) * H_ + col] = (_Float16)(og * tanh_(c0r[r]));
    }

    grid_barrier(barc, (unsigned)(t + 1));   // single barrier per step

    // -------- layer 1: gates = b1 + h0new @ Wi1^T + h1 @ Wh1^T --------
#pragma unroll
    for (int g = 0; g < 4; ++g)
#pragma unroll
      for (int r = 0; r < 8; ++r) acc[g][r] = z1b[g];
    mm4_ldsB(acc, h0n, Bsm[1], mtile, lane);
    mm4_ldsB(acc, h1c, Bsm[2], mtile, lane);

#pragma unroll
    for (int r = 0; r < 8; ++r) {
      const float ig = sigm_(acc[0][r]);
      const float fg = sigm_(acc[1][r]);
      const float gg = tanh_(acc[2][r]);
      const float og = sigm_(acc[3][r]);
      c1r[r] = fg * c1r[r] + ig * gg;
      const float hn = og * tanh_(c1r[r]);
      const int row  = rbase + r;
      h1n[row * H_ + col] = (_Float16)hn;
      seq[(row * T_ + t) * H_ + col] = (_Float16)hn;   // seq[b][t][h]
    }

    // ping-pong (no trailing barrier needed: next step's barrier orders p1->p1)
    _Float16* tp;
    tp = h0c; h0c = h0n; h0n = tp;
    tp = h1c; h1c = h1n; h1n = tp;
  }
}

// ---------------------------------------------------------------- FC + softmax

__global__ __launch_bounds__(TPBF) void fc_softmax_kernel(
    const _Float16* __restrict__ seq, const _Float16* __restrict__ Wf,
    const float* __restrict__ fcb, float* __restrict__ out) {
  const int w    = blockIdx.x * (TPBF / 32) + (threadIdx.x >> 5);  // m-tile 0..8191
  const int lane = threadIdx.x & 31;
  const int nlo  = lane & 15;
  const int hi   = lane >> 4;

  v8f acc[16];
#pragma unroll
  for (int nt = 0; nt < 16; ++nt)
#pragma unroll
    for (int r = 0; r < 8; ++r) acc[nt][r] = 0.0f;

  const _Float16* Ar = seq + (size_t)(w * 16 + nlo) * H_;
  __builtin_prefetch(Ar, 0, 1);
#pragma unroll
  for (int kb = 0; kb < H_ / 32; ++kb) {
    const int k0 = kb * 32;
    v16h a = load_a_frag(Ar, k0, hi);
    const int kB = k0 + hi * 16;
#pragma unroll
    for (int nt = 0; nt < 16; ++nt) {
      v16h b = *(const v16h*)(Wf + (nt * 16 + nlo) * H_ + kB);
      acc[nt] = wmma16(a, b, acc[nt]);
    }
  }

  float fb[16];
#pragma unroll
  for (int nt = 0; nt < 16; ++nt) fb[nt] = fcb[nt * 16 + nlo];

  const size_t HALF = (size_t)B_ * T_ * O_;
#pragma unroll
  for (int r = 0; r < 8; ++r) {
    float mx = -3.0e38f;
#pragma unroll
    for (int nt = 0; nt < 16; ++nt) {
      acc[nt][r] += fb[nt];
      mx = fmaxf(mx, acc[nt][r]);
    }
#pragma unroll
    for (int m = 1; m < 16; m <<= 1) mx = fmaxf(mx, __shfl_xor(mx, m, 32));
    float s = 0.0f;
#pragma unroll
    for (int nt = 0; nt < 16; ++nt) s += __expf(acc[nt][r] - mx);
#pragma unroll
    for (int m = 1; m < 16; m <<= 1) s += __shfl_xor(s, m, 32);
    const float ls = __logf(s);

    const size_t base = (size_t)(w * 16 + r + hi * 8) * O_ + nlo;
#pragma unroll
    for (int nt = 0; nt < 16; ++nt) {
      const float v = acc[nt][r] - mx - ls;     // log_softmax
      out[base + nt * 16]        = v;
      out[HALF + base + nt * 16] = __expf(v);   // softmax = exp(log_softmax)
    }
  }
}

// ---------------------------------------------------------------- launch

extern "C" void kernel_launch(void* const* d_in, const int* in_sizes, int n_in,
                              void* d_out, int out_size, void* d_ws, size_t ws_size,
                              hipStream_t stream) {
  const float* x   = (const float*)d_in[0];   // (256,1,256)
  const float* h0  = (const float*)d_in[1];   // (2,256,256)
  const float* c0i = (const float*)d_in[2];   // (2,256,256)
  const float* Wih = (const float*)d_in[3];   // (2,1024,256)
  const float* Whh = (const float*)d_in[4];   // (2,1024,256)
  const float* bih = (const float*)d_in[5];   // (2,1024)
  const float* bhh = (const float*)d_in[6];   // (2,1024)
  const float* fcw = (const float*)d_in[7];   // (256,256)
  const float* fcb = (const float*)d_in[8];   // (256,)
  (void)in_sizes; (void)n_in; (void)out_size; (void)ws_size;

  const int WSZ = 4 * H_ * H_;   // 262144
  const int HSZ = B_ * H_;       // 65536

  char* p = (char*)d_ws;
  auto take = [&](size_t bytes) -> void* {
    void* r = (void*)p;
    p += (bytes + 255) & ~(size_t)255;
    return r;
  };
  _Float16* Wi0h = (_Float16*)take((size_t)WSZ * 2);
  _Float16* Wh0h = (_Float16*)take((size_t)WSZ * 2);
  _Float16* Wi1h = (_Float16*)take((size_t)WSZ * 2);
  _Float16* Wh1h = (_Float16*)take((size_t)WSZ * 2);
  _Float16* fcwh = (_Float16*)take((size_t)O_ * H_ * 2);
  _Float16* xh   = (_Float16*)take((size_t)HSZ * 2);
  _Float16* h0a  = (_Float16*)take((size_t)HSZ * 2);
  _Float16* h0b  = (_Float16*)take((size_t)HSZ * 2);
  _Float16* h1a  = (_Float16*)take((size_t)HSZ * 2);
  _Float16* h1b  = (_Float16*)take((size_t)HSZ * 2);
  unsigned* barc = (unsigned*)take(256);
  _Float16* seq  = (_Float16*)take((size_t)B_ * T_ * H_ * 2);   // 64 MB

  const int cb = 256;
  cvt_f16_kernel<<<(WSZ + cb - 1) / cb, cb, 0, stream>>>(Wi0h, Wih, WSZ);
  cvt_f16_kernel<<<(WSZ + cb - 1) / cb, cb, 0, stream>>>(Wh0h, Whh, WSZ);
  cvt_f16_kernel<<<(WSZ + cb - 1) / cb, cb, 0, stream>>>(Wi1h, Wih + WSZ, WSZ);
  cvt_f16_kernel<<<(WSZ + cb - 1) / cb, cb, 0, stream>>>(Wh1h, Whh + WSZ, WSZ);
  cvt_f16_kernel<<<(O_ * H_ + cb - 1) / cb, cb, 0, stream>>>(fcwh, fcw, O_ * H_);
  cvt_f16_kernel<<<(HSZ + cb - 1) / cb, cb, 0, stream>>>(xh, x, HSZ);
  cvt_f16_kernel<<<(HSZ + cb - 1) / cb, cb, 0, stream>>>(h0a, h0, HSZ);
  cvt_f16_kernel<<<(HSZ + cb - 1) / cb, cb, 0, stream>>>(h1a, h0 + HSZ, HSZ);
  init_barrier_kernel<<<1, 64, 0, stream>>>(barc);

  lstm_kernel<<<NBLK, TPBL, 0, stream>>>(xh, Wi0h, Wh0h, Wi1h, Wh1h,
                                         bih, bhh, c0i, c0i + HSZ,
                                         h0a, h0b, h1a, h1b, seq, barc);

  fc_softmax_kernel<<<(B_ * T_ / 16) / (TPBF / 32), TPBF, 0, stream>>>(
      seq, fcwh, fcb, (float*)d_out);
}
